// SSMBlock_1975684956480
// MI455X (gfx1250) — matrix operand order repaired and checked by reading, not compiled, not run on hardware
//
#include <hip/hip_runtime.h>
#include <hip/hip_bf16.h>

typedef __attribute__((ext_vector_type(16))) __bf16 v16bf;
typedef __attribute__((ext_vector_type(8)))  __bf16 v8bf;
typedef __attribute__((ext_vector_type(8)))  float  v8f;
typedef __attribute__((ext_vector_type(4))) unsigned int u32x4;
typedef __attribute__((ext_vector_type(8)))  int i32x8;
typedef __attribute__((ext_vector_type(4)))  int i32x4;

#define D_MODEL 768
#define D_STATE 192
#define BATCH   4
#define SEQ     4096
#define MTOT    (BATCH*SEQ)   /* 16384 rows */
#define CHUNK   64
#define NCHUNK  (SEQ/CHUNK)   /* 64 chunks */

// ---------------------------------------------------------------------------
// Tensor Data Mover: async 2-D tile load (bf16 elements) global -> LDS.
// D# packed per CDNA5 ISA 8.3/8.4: group0 = {flags, lds_addr, global_addr,
// type=2}; group1 = {data_size=2B, tensor_dim0/1 (OOB zero-fill), tile_dim0/1,
// dim0 stride}; groups 2/3 zero (2-D tensor). Tracked by TENSORcnt.
// This toolchain exposes the 6-arg builtin (extra i32x8 group + cpol).
// ---------------------------------------------------------------------------
__device__ __forceinline__ void tdm_load_2d(unsigned lds_off, const void* gptr,
                                            unsigned tile_x, unsigned tile_y,
                                            unsigned tensor_x, unsigned tensor_y,
                                            unsigned stride_elems)
{
    unsigned long long ga = (unsigned long long)gptr;
    u32x4 g0;
    g0[0] = 1u;                                   // count=1, user descriptor
    g0[1] = lds_off;                              // lds_addr (bytes)
    g0[2] = (unsigned)ga;                         // global_addr[31:0]
    g0[3] = (unsigned)((ga >> 32) & 0x01ffffffu) | (2u << 30); // addr[56:32]|type=2
    i32x8 g1;
    g1[0] = (int)(1u << 16);                      // wg_mask=0, data_size=1 (2B)
    g1[1] = (int)((tensor_x & 0xffffu) << 16);    // tensor_dim0[15:0] @ [63:48]
    g1[2] = (int)(((tensor_x >> 16) & 0xffffu) | ((tensor_y & 0xffffu) << 16));
    g1[3] = (int)(((tensor_y >> 16) & 0xffffu) | ((tile_x & 0xffffu) << 16));
    g1[4] = (int)(tile_y & 0xffffu);              // tile_dim1 (tile_dim2=0)
    g1[5] = (int)stride_elems;                    // tensor_dim0_stride[31:0]
    g1[6] = 0;
    g1[7] = 0;
    i32x4 gz4 = (i32x4){0, 0, 0, 0};
    i32x8 gz8 = (i32x8){0, 0, 0, 0, 0, 0, 0, 0};
    __builtin_amdgcn_tensor_load_to_lds(g0, g1, gz4, gz4, gz8, 0);
}

// ---------------------------------------------------------------------------
// Elementwise f32 -> bf16 convert
// ---------------------------------------------------------------------------
__global__ void cvt_bf(const float* __restrict__ in, __bf16* __restrict__ out, long n) {
    long i = (long)blockIdx.x * blockDim.x + threadIdx.x;
    if (i < n) out[i] = (__bf16)in[i];
}

// Transpose-convert: in f32 [R,C] -> out_t bf16 [C,R] (and optional plain copy)
__global__ void transcvt(const float* __restrict__ in, int R, int C,
                         __bf16* __restrict__ out_t, __bf16* __restrict__ out_n) {
    long i = (long)blockIdx.x * blockDim.x + threadIdx.x;
    long total = (long)R * C;
    if (i >= total) return;
    int r = (int)(i / C), c = (int)(i % C);
    float v = in[i];
    if (out_t) out_t[(long)c * R + r] = (__bf16)v;
    if (out_n) out_n[i] = (__bf16)v;
}

// ---------------------------------------------------------------------------
// bf16 WMMA GEMM:  D[M,N] = Abf[M,K] * Bbf[N,K]^T.
// 256 thr = 8 waves; block tile 128x128, wave strip 16(M) x 128(N).
// A/B 128x32 tiles staged into LDS by the Tensor Data Mover (wave 0 issues,
// s_wait_tensorcnt + barrier), double-buffered so DMA overlaps the 8 WMMAs.
// TDM OOB zero-fill handles ragged M/N edges (stores still guarded).
// MODE 0: plain f32   MODE 1: +bias, sigmoid->gate_bf / xin_bf split
// MODE 2: y=acc+xin*D, z=y*gate -> z_bf   MODE 3: +bias -> f32 out
// ---------------------------------------------------------------------------
template<int MODE>
__global__ void gemm_wmma(const __bf16* __restrict__ Abf,
                          const __bf16* __restrict__ Bbf,
                          int M, int N, int K,
                          const float*  __restrict__ bias,
                          float*        __restrict__ out_f32,
                          __bf16*       __restrict__ gate_bf,
                          __bf16*       __restrict__ xin_bf,
                          const float*  __restrict__ Dvec,
                          __bf16*       __restrict__ z_bf)
{
    extern __shared__ char smem[];                 // 32KB: A0,B0,A1,B1 (8KB each)
    const int lane  = threadIdx.x & 31;
    const int wave  = threadIdx.x >> 5;
    const int hi    = lane >> 4;
    const int l16   = lane & 15;
    const int mrow0 = blockIdx.y * 128;
    const int m0    = mrow0 + wave * 16;
    const int nbase = blockIdx.x * 128;
    const int nk    = K / 32;
    const unsigned Mrem = (unsigned)(M - mrow0);   // TDM zero-fills rows >= Mrem
    const unsigned Nrem = (unsigned)(N - nbase);
    const unsigned aoff[2] = {0u, 16384u};
    const unsigned boff[2] = {8192u, 24576u};

    if (wave == 0) {
        tdm_load_2d(aoff[0], Abf + (long)mrow0 * K, 32u, 128u, (unsigned)K, Mrem, (unsigned)K);
        tdm_load_2d(boff[0], Bbf + (long)nbase * K, 32u, 128u, (unsigned)K, Nrem, (unsigned)K);
        __builtin_amdgcn_s_wait_tensorcnt(0);
    }
    __syncthreads();

    v8f acc[8];
    #pragma unroll
    for (int t = 0; t < 8; t++) { v8f z = {}; acc[t] = z; }

    for (int it = 0; it < nk; ++it) {
        const int cur = it & 1, nxt = cur ^ 1;
        if (wave == 0 && it + 1 < nk) {            // prefetch next K-slab via TDM
            const int kb2 = (it + 1) * 32;
            tdm_load_2d(aoff[nxt], Abf + (long)mrow0 * K + kb2, 32u, 128u, (unsigned)K, Mrem, (unsigned)K);
            tdm_load_2d(boff[nxt], Bbf + (long)nbase * K + kb2, 32u, 128u, (unsigned)K, Nrem, (unsigned)K);
        }
        const __bf16* As = (const __bf16*)(smem + aoff[cur]);  // [128][32]
        const __bf16* Bs = (const __bf16*)(smem + boff[cur]);  // [128][32]

        const __bf16* ap = As + (wave * 16 + l16) * 32;
        v8bf alo = *(const v8bf*)(ap + (hi ? 16 : 0));
        v8bf ahi = *(const v8bf*)(ap + (hi ? 24 : 8));
        v16bf a  = __builtin_shufflevector(alo, ahi,
                     0,1,2,3,4,5,6,7,8,9,10,11,12,13,14,15);
        #pragma unroll
        for (int nt = 0; nt < 8; ++nt) {
            const __bf16* bp = Bs + (nt * 16 + l16) * 32 + hi * 16;
            v8bf b0 = *(const v8bf*)bp;
            v8bf b1 = *(const v8bf*)(bp + 8);
            v16bf b = __builtin_shufflevector(b0, b1,
                        0,1,2,3,4,5,6,7,8,9,10,11,12,13,14,15);
            acc[nt] = __builtin_amdgcn_wmma_f32_16x16x32_bf16(
                false, a, false, b, (short)0, acc[nt], false, false);
        }
        if (wave == 0 && it + 1 < nk) __builtin_amdgcn_s_wait_tensorcnt(0);
        __syncthreads();
    }

    if (m0 < M) {
        const int rowoff = hi * 8;
        #pragma unroll
        for (int nt = 0; nt < 8; nt++) {
            int n0 = nbase + nt * 16;
            if (n0 >= N) continue;
            int n = n0 + l16;
            #pragma unroll
            for (int r = 0; r < 8; r++) {
                int   m = m0 + rowoff + r;
                float v = acc[nt][r];
                if (MODE == 0) {
                    out_f32[(long)m * N + n] = v;
                } else if (MODE == 1) {
                    v += bias[n];
                    if (n < D_MODEL)
                        gate_bf[(long)m * D_MODEL + n] = (__bf16)(1.0f / (1.0f + __expf(-v)));
                    else
                        xin_bf[(long)m * D_MODEL + (n - D_MODEL)] = (__bf16)v;
                } else if (MODE == 2) {
                    float y = v + (float)xin_bf[(long)m * D_MODEL + n] * Dvec[n];
                    float z = y * (float)gate_bf[(long)m * D_MODEL + n];
                    z_bf[(long)m * D_MODEL + n] = (__bf16)z;
                } else {
                    out_f32[(long)m * N + n] = v + bias[n];
                }
            }
        }
    }
}

// ---------------------------------------------------------------------------
// WMMA scan step kernel:  h_t = h_{t-1} * Amat^T + u_t   (Amat as [N,K] B-op)
// 12 waves (384 thr), wave w owns state columns [16w,16w+16). A-matrix lives in
// registers for the whole chunk; h is kept in LDS as bf16 (it is rounded to
// bf16 for the WMMA anyway), so each K-chunk fragment is just 2 ds_load_b128.
// u_t is injected via the WMMA C operand. Generic strides let the same kernel
// run pass A (chunk-local), pass B (combine with A^64), pass C (bf16 states).
// ---------------------------------------------------------------------------
__global__ void scan_wmma(const __bf16* __restrict__ Amat,               // [192,192]
                          const float*  __restrict__ u,  long ucs, long uts, long ubs,
                          const float*  __restrict__ hinit, long hics,
                          float*        __restrict__ opre,  long pcs, long pts, long pbs,
                          __bf16*       __restrict__ opost, long qcs, long qts, long qbs,
                          float*        __restrict__ ofin,  long fcs,
                          int steps)
{
    __shared__ __bf16 hbuf[2][16][D_STATE];
    const int tid  = threadIdx.x;
    const int lane = tid & 31;
    const int wave = tid >> 5;              // 0..11
    const int chunk = blockIdx.x;
    const float* up = u + (long)chunk * ucs;

    for (int i = tid; i < 2 * 16 * D_STATE; i += blockDim.x)
        ((__bf16*)hbuf)[i] = (__bf16)0.0f;
    __syncthreads();
    if (hinit) {
        const float* hp = hinit + (long)chunk * hics;
        for (int i = tid; i < BATCH * D_STATE; i += blockDim.x)
            hbuf[0][i / D_STATE][i % D_STATE] = (__bf16)hp[i];
    }
    __syncthreads();

    const int hi   = lane >> 4;
    const int nrow = wave * 16 + (lane & 15);
    v16bf bfrag[6];
    #pragma unroll
    for (int kk = 0; kk < 6; kk++)
        bfrag[kk] = *(const v16bf*)(Amat + (long)nrow * D_STATE + kk * 32 + hi * 16);

    const int col  = wave * 16 + (lane & 15);
    const int arow = lane & 15;

    for (int t = 0; t < steps; t++) {
        const int cur = t & 1, nxt = cur ^ 1;

        if (opre) {  // record pre-step state (pass B -> chunk start states)
            float* pp = opre + (long)chunk * pcs + (long)t * pts;
            for (int i = tid; i < BATCH * D_STATE; i += blockDim.x)
                pp[(long)(i / D_STATE) * pbs + (i % D_STATE)] =
                    (float)hbuf[cur][i / D_STATE][i % D_STATE];
        }

        // C operand seeded with u_t (rows 0..3 = elements 0..3 of lanes<16)
        v8f acc = {};
        if (hi == 0) {
            #pragma unroll
            for (int r = 0; r < 4; r++)
                acc[r] = up[(long)r * ubs + (long)t * uts + col];
        }

        #pragma unroll
        for (int kk = 0; kk < 6; kk++) {
            const __bf16* hrow = &hbuf[cur][arow][kk * 32];
            v8bf lo = *(const v8bf*)(hrow + (hi ? 16 : 0));
            v8bf hh = *(const v8bf*)(hrow + (hi ? 24 : 8));
            v16bf a = __builtin_shufflevector(lo, hh,
                        0,1,2,3,4,5,6,7,8,9,10,11,12,13,14,15);
            acc = __builtin_amdgcn_wmma_f32_16x16x32_bf16(
                false, a, false, bfrag[kk], (short)0, acc, false, false);
        }

        if (hi == 0) {
            #pragma unroll
            for (int r = 0; r < 4; r++) hbuf[nxt][r][col] = (__bf16)acc[r];
            if (opost) {
                __bf16* qp = opost + (long)chunk * qcs + (long)t * qts;
                #pragma unroll
                for (int r = 0; r < 4; r++)
                    qp[(long)r * qbs + col] = (__bf16)acc[r];
            }
        }
        __syncthreads();
    }

    if (ofin) {
        float* fp = ofin + (long)chunk * fcs;
        const int last = steps & 1;
        for (int i = tid; i < BATCH * D_STATE; i += blockDim.x)
            fp[i] = (float)hbuf[last][i / D_STATE][i % D_STATE];
    }
}

// ---------------------------------------------------------------------------
extern "C" void kernel_launch(void* const* d_in, const int* in_sizes, int n_in,
                              void* d_out, int out_size, void* d_ws, size_t ws_size,
                              hipStream_t stream) {
    const float* x    = (const float*)d_in[0];
    const float* Amat = (const float*)d_in[1];
    const float* Bm   = (const float*)d_in[2];
    const float* Cm   = (const float*)d_in[3];
    const float* Dv   = (const float*)d_in[4];
    const float* Wi   = (const float*)d_in[5];
    const float* bi   = (const float*)d_in[6];
    const float* Wo   = (const float*)d_in[7];
    const float* bo   = (const float*)d_in[8];

    char* ws = (char*)d_ws;
    size_t off = 0;
    auto alloc = [&](size_t bytes) -> char* {
        char* p = ws + off;
        off = (off + bytes + 255) & ~(size_t)255;
        return p;
    };
    __bf16* xbf    = (__bf16*)alloc((size_t)MTOT * D_MODEL * 2);
    __bf16* Wibf   = (__bf16*)alloc((size_t)2 * D_MODEL * D_MODEL * 2);
    __bf16* Wobf   = (__bf16*)alloc((size_t)D_MODEL * D_MODEL * 2);
    __bf16* Abf    = (__bf16*)alloc((size_t)D_STATE * D_STATE * 2);
    __bf16* ATbf   = (__bf16*)alloc((size_t)D_STATE * D_STATE * 2);
    __bf16* BmTbf  = (__bf16*)alloc((size_t)D_STATE * D_MODEL * 2);
    __bf16* CTbf   = (__bf16*)alloc((size_t)D_MODEL * D_STATE * 2);
    __bf16* gatebf = (__bf16*)alloc((size_t)MTOT * D_MODEL * 2);
    __bf16* xinbf  = (__bf16*)alloc((size_t)MTOT * D_MODEL * 2);
    float*  u32    = (float*) alloc((size_t)MTOT * D_STATE * 4);
    __bf16* Pbuf0  = (__bf16*)alloc((size_t)D_STATE * D_STATE * 2);
    __bf16* Pbuf1  = (__bf16*)alloc((size_t)D_STATE * D_STATE * 2);
    __bf16* PTbuf0 = (__bf16*)alloc((size_t)D_STATE * D_STATE * 2);
    __bf16* PTbuf1 = (__bf16*)alloc((size_t)D_STATE * D_STATE * 2);
    float*  Psq    = (float*) alloc((size_t)D_STATE * D_STATE * 4);
    float*  vbuf   = (float*) alloc((size_t)NCHUNK * BATCH * D_STATE * 4);
    float*  hstart = (float*) alloc((size_t)NCHUNK * BATCH * D_STATE * 4);
    __bf16* hsbf   = (__bf16*)alloc((size_t)MTOT * D_STATE * 2);
    __bf16* zbf    = (__bf16*)alloc((size_t)MTOT * D_MODEL * 2);

    const dim3 blk(256);
    const size_t GEMM_LDS = 32768;
    auto g1 = [](long n) { return dim3((unsigned)((n + 255) / 256)); };

    // ---- precision staging (bf16 operands; weights in [N,K] form) ----
    cvt_bf<<<g1((long)MTOT * D_MODEL), blk, 0, stream>>>(x, xbf, (long)MTOT * D_MODEL);
    cvt_bf<<<g1((long)2 * D_MODEL * D_MODEL), blk, 0, stream>>>(Wi, Wibf, (long)2 * D_MODEL * D_MODEL);
    cvt_bf<<<g1((long)D_MODEL * D_MODEL), blk, 0, stream>>>(Wo, Wobf, (long)D_MODEL * D_MODEL);
    cvt_bf<<<g1((long)D_STATE * D_STATE), blk, 0, stream>>>(Amat, Abf, (long)D_STATE * D_STATE);
    transcvt<<<g1((long)D_MODEL * D_STATE), blk, 0, stream>>>(Bm, D_MODEL, D_STATE, BmTbf, nullptr);
    transcvt<<<g1((long)D_STATE * D_MODEL), blk, 0, stream>>>(Cm, D_STATE, D_MODEL, CTbf, nullptr);
    transcvt<<<g1((long)D_STATE * D_STATE), blk, 0, stream>>>(Amat, D_STATE, D_STATE, ATbf, nullptr);

    // ---- fused input projection: gate + x_input ; then u = x_input*Bm ----
    gemm_wmma<1><<<dim3(12, MTOT / 128), blk, GEMM_LDS, stream>>>(
        xbf, Wibf, MTOT, 2 * D_MODEL, D_MODEL, bi, nullptr, gatebf, xinbf, nullptr, nullptr);
    gemm_wmma<0><<<dim3(2, MTOT / 128), blk, GEMM_LDS, stream>>>(
        xinbf, BmTbf, MTOT, D_STATE, D_MODEL, nullptr, u32, nullptr, nullptr, nullptr, nullptr);

    // ---- A^64 by repeated squaring (P and P^T kept so B-operand is [N,K]) ----
    const __bf16* Pc = Abf; const __bf16* PTc = ATbf;
    __bf16* Pn[2]  = { Pbuf0, Pbuf1 };
    __bf16* PTn[2] = { PTbuf0, PTbuf1 };
    for (int i = 0; i < 6; i++) {
        gemm_wmma<0><<<dim3(2, 2), blk, GEMM_LDS, stream>>>(
            (const __bf16*)Pc, (const __bf16*)PTc, D_STATE, D_STATE, D_STATE,
            nullptr, Psq, nullptr, nullptr, nullptr, nullptr);
        transcvt<<<g1((long)D_STATE * D_STATE), blk, 0, stream>>>(
            Psq, D_STATE, D_STATE, PTn[i & 1], Pn[i & 1]);
        Pc = Pn[i & 1]; PTc = PTn[i & 1];
    }
    const __bf16* A64bf = Pc;

    // ---- chunked scan: A (local), B (combine with A^64), C (final, bf16) ----
    scan_wmma<<<NCHUNK, 384, 0, stream>>>(
        Abf, u32, (long)CHUNK * D_STATE, D_STATE, (long)SEQ * D_STATE,
        nullptr, 0,
        nullptr, 0, 0, 0,
        nullptr, 0, 0, 0,
        vbuf, (long)BATCH * D_STATE, CHUNK);
    scan_wmma<<<1, 384, 0, stream>>>(
        A64bf, vbuf, 0, (long)BATCH * D_STATE, D_STATE,
        nullptr, 0,
        hstart, 0, (long)BATCH * D_STATE, D_STATE,
        nullptr, 0, 0, 0,
        nullptr, 0, NCHUNK);
    scan_wmma<<<NCHUNK, 384, 0, stream>>>(
        Abf, u32, (long)CHUNK * D_STATE, D_STATE, (long)SEQ * D_STATE,
        hstart, (long)BATCH * D_STATE,
        nullptr, 0, 0, 0,
        hsbf, (long)CHUNK * D_STATE, D_STATE, (long)SEQ * D_STATE,
        nullptr, 0, CHUNK);

    // ---- y = hs*C + x_input*D, z = y*gate ; out = z*W_out^T + b_out ----
    gemm_wmma<2><<<dim3(6, MTOT / 128), blk, GEMM_LDS, stream>>>(
        hsbf, CTbf, MTOT, D_MODEL, D_STATE, nullptr, nullptr, gatebf, xinbf, Dv, zbf);
    gemm_wmma<3><<<dim3(6, MTOT / 128), blk, GEMM_LDS, stream>>>(
        zbf, Wobf, MTOT, D_MODEL, D_MODEL, bo, (float*)d_out, nullptr, nullptr, nullptr, nullptr);
}
